// GNNModel_9345848836715
// MI455X (gfx1250) — matrix operand order repaired and checked by reading, not compiled, not run on hardware
//
#include <hip/hip_runtime.h>
#include <hip/hip_bf16.h>

typedef __attribute__((ext_vector_type(16))) __bf16 v16bf;
typedef __attribute__((ext_vector_type(8)))  float  v8f;

#define NU      50000
#define NB      50000
#define NNODES  100000
#define NEDGES  1250000
#define FIN     512
#define HDIM    64
#define BN_EPS  1e-5f

#if __has_builtin(__builtin_amdgcn_sched_barrier)
#define SCHED_FENCE() __builtin_amdgcn_sched_barrier(0)
#else
#define SCHED_FENCE()
#endif

// ---------- helpers ----------
// Native f32->bf16 (fptrunc is round-to-nearest-even; lowers to hardware cvt on gfx1250,
// pairs pattern-match to v_cvt_pk_bf16_f32 where available).
__device__ __forceinline__ unsigned int pack2_bf16(float lo, float hi) {
    union { __bf16 b[2]; unsigned int u; } c;
    c.b[0] = (__bf16)lo;
    c.b[1] = (__bf16)hi;
    return c.u;
}

union FragU { v16bf v; uint4 q[2]; };

// ---------- weight pre-pack: W[64,K] f32 -> per-lane B-fragment bf16 tiles ----------
// B[k,n] = W[n,k]. Tile (kt,nt): lane holds n=16*nt+(lane&15), kh=lane>>4,
// dword v = pack(W[n][kt*32+16*kh+2v], W[n][kt*32+16*kh+2v+1]); 8 dwords/lane.
__global__ __launch_bounds__(256)
void pack_weight(const float* __restrict__ W, unsigned int* __restrict__ Wp, int K) {
    const int lane = threadIdx.x & 31;
    const int wave = threadIdx.x >> 5;
    const int tile = blockIdx.x * (blockDim.x >> 5) + wave;
    const int Kt = K >> 5;
    if (tile >= 4 * Kt) return;
    const int nt = tile / Kt, kt = tile % Kt;
    const int n  = nt * 16 + (lane & 15);
    const int kh = lane >> 4;
    const float4* p = (const float4*)(W + (size_t)n * K + kt * 32 + kh * 16);
    float4 w0 = p[0], w1 = p[1], w2 = p[2], w3 = p[3];
    uint4 q0 = make_uint4(pack2_bf16(w0.x, w0.y), pack2_bf16(w0.z, w0.w),
                          pack2_bf16(w1.x, w1.y), pack2_bf16(w1.z, w1.w));
    uint4 q1 = make_uint4(pack2_bf16(w2.x, w2.y), pack2_bf16(w2.z, w2.w),
                          pack2_bf16(w3.x, w3.y), pack2_bf16(w3.z, w3.w));
    uint4* dst = (uint4*)Wp + ((size_t)tile * 32 + lane) * 2;
    dst[0] = q0;
    dst[1] = q1;
}

// raw A (4x float4) + 4 B fragments for one K-step
__device__ __forceinline__ void load_step(const float* __restrict__ xrow,
                                          const uint4* __restrict__ bwave, size_t ntstep,
                                          int kt, int h, float4 a[4], FragU B[4]) {
    const float* p = xrow + kt * 32 + 16 * h;
    a[0] = *(const float4*)(p);
    a[1] = *(const float4*)(p + 4);
    a[2] = *(const float4*)(p + 8);
    a[3] = *(const float4*)(p + 12);
    const uint4* bk = bwave + (size_t)kt * 64;
    B[0].q[0] = bk[0];               B[0].q[1] = bk[1];
    B[1].q[0] = bk[ntstep];          B[1].q[1] = bk[ntstep + 1];
    B[2].q[0] = bk[2 * ntstep];      B[2].q[1] = bk[2 * ntstep + 1];
    B[3].q[0] = bk[3 * ntstep];      B[3].q[1] = bk[3 * ntstep + 1];
}

__device__ __forceinline__ void pack_afrag(const float4 a[4], FragU& A) {
    A.q[0] = make_uint4(pack2_bf16(a[0].x, a[0].y), pack2_bf16(a[0].z, a[0].w),
                        pack2_bf16(a[1].x, a[1].y), pack2_bf16(a[1].z, a[1].w));
    A.q[1] = make_uint4(pack2_bf16(a[2].x, a[2].y), pack2_bf16(a[2].z, a[2].w),
                        pack2_bf16(a[3].x, a[3].y), pack2_bf16(a[3].z, a[3].w));
}

// ---------- fused GEMM: out[M,64] = epilogue( X[M,K] @ W^T ) via V_WMMA_F32_16X16X32_BF16 ----------
// One wave per 16-row tile; 4 N-tiles of 16 cols. All guards wave-uniform (EXEC all-ones at WMMA).
// K-loop unrolled by 2 with ping-pong buffers (Kt is always even: 2 or 16): compute on buffer A,
// refill buffer A for step kt+2, compute on buffer B, refill buffer B for step kt+3. No register
// rotation, no branches in the body; last-iteration prefetch index clamped (redundant L2 hit).
__global__ __launch_bounds__(256)
void gemm_wmma_bf16(const float* __restrict__ X, const unsigned int* __restrict__ Wp,
                    float* __restrict__ out, int M, int K,
                    const float* __restrict__ rowscale,   // per-row (inv-degree) or null
                    const float* __restrict__ bias,       // per-channel [64] or null
                    const float* __restrict__ addsrc,     // [M,64] residual or null
                    const float* __restrict__ bn_g, const float* __restrict__ bn_b,
                    const float* __restrict__ bn_m, const float* __restrict__ bn_v,
                    int relu) {
    const int lane = threadIdx.x & 31;
    const int wave = threadIdx.x >> 5;
    const int tile = blockIdx.x * (blockDim.x >> 5) + wave;
    const int mrow = tile * 16;
    if (mrow >= M) return;                      // wave-uniform
    const int Kt = K >> 5;                      // always even (2 or 16)
    const int m  = lane & 15;
    const int h  = lane >> 4;
    const float* xrow = X + (size_t)(mrow + m) * K;
    const uint4* bwave = (const uint4*)Wp + (size_t)lane * 2;   // + (nt*Kt + kt)*64
    const size_t ntstep = (size_t)Kt * 64;                      // uint4 stride between N-tiles

    v8f acc0 = {}, acc1 = {}, acc2 = {}, acc3 = {};
    float4 aA[4], aB[4];
    FragU  BA[4], BB[4];
    load_step(xrow, bwave, ntstep, 0, h, aA, BA);
    load_step(xrow, bwave, ntstep, 1, h, aB, BB);
    for (int kt = 0; kt < Kt; kt += 2) {
        FragU A0;
        pack_afrag(aA, A0);
        acc0 = __builtin_amdgcn_wmma_f32_16x16x32_bf16(false, A0.v, false, BA[0].v, (short)0, acc0, false, false);
        acc1 = __builtin_amdgcn_wmma_f32_16x16x32_bf16(false, A0.v, false, BA[1].v, (short)0, acc1, false, false);
        acc2 = __builtin_amdgcn_wmma_f32_16x16x32_bf16(false, A0.v, false, BA[2].v, (short)0, acc2, false, false);
        acc3 = __builtin_amdgcn_wmma_f32_16x16x32_bf16(false, A0.v, false, BA[3].v, (short)0, acc3, false, false);
        SCHED_FENCE();
        const int k2 = (kt + 2 < Kt) ? kt + 2 : Kt - 1;   // clamped prefetch (no branch)
        load_step(xrow, bwave, ntstep, k2, h, aA, BA);
        FragU A1;
        pack_afrag(aB, A1);
        acc0 = __builtin_amdgcn_wmma_f32_16x16x32_bf16(false, A1.v, false, BB[0].v, (short)0, acc0, false, false);
        acc1 = __builtin_amdgcn_wmma_f32_16x16x32_bf16(false, A1.v, false, BB[1].v, (short)0, acc1, false, false);
        acc2 = __builtin_amdgcn_wmma_f32_16x16x32_bf16(false, A1.v, false, BB[2].v, (short)0, acc2, false, false);
        acc3 = __builtin_amdgcn_wmma_f32_16x16x32_bf16(false, A1.v, false, BB[3].v, (short)0, acc3, false, false);
        SCHED_FENCE();
        const int k3 = (kt + 3 < Kt) ? kt + 3 : Kt - 1;
        load_step(xrow, bwave, ntstep, k3, h, aB, BB);
    }
    v8f accs[4] = {acc0, acc1, acc2, acc3};

    // Epilogue. D layout: lane -> col n = 16*nt + (lane&15); vgpr j -> row mrow + j + 8*(lane>>4)
    const int nl    = lane & 15;
    const int mbase = mrow + 8 * (lane >> 4);
#pragma unroll
    for (int nt = 0; nt < 4; ++nt) {
        const int n = nt * 16 + nl;
        float b  = bias ? bias[n] : 0.0f;
        float sc = 1.0f, sh = 0.0f, mu = 0.0f;
        if (bn_g) {
            sc = bn_g[n] * rsqrtf(bn_v[n] + BN_EPS);
            mu = bn_m[n];
            sh = bn_b[n];
        }
#pragma unroll
        for (int j = 0; j < 8; ++j) {
            const int mr = mbase + j;
            float v = accs[nt][j];
            if (rowscale) v *= rowscale[mr];
            v += b;
            if (addsrc) v += addsrc[(size_t)mr * HDIM + n];
            if (bn_g)   v = (v - mu) * sc + sh;
            if (relu)   v = fmaxf(v, 0.0f);
            out[(size_t)mr * HDIM + n] = v;
        }
    }
}

// ---------- graph kernels ----------
__global__ __launch_bounds__(256)
void zero4_kernel(float4* __restrict__ p, int n4) {
    int i = blockIdx.x * blockDim.x + threadIdx.x;
    if (i < n4) p[i] = make_float4(0.f, 0.f, 0.f, 0.f);
}

__global__ __launch_bounds__(256)
void degree_kernel(const int* __restrict__ dst, float* __restrict__ deg, int E) {
    int e = blockIdx.x * blockDim.x + threadIdx.x;
    if (e < E) atomicAdd(&deg[dst[e]], 1.0f);
}

__global__ __launch_bounds__(256)
void invdeg_kernel(float* __restrict__ deg, int n) {
    int i = blockIdx.x * blockDim.x + threadIdx.x;
    if (i < n) deg[i] = 1.0f / fmaxf(deg[i], 1.0f);
}

// thread = (edge, channel-pair); a wave covers all 64 channels of one edge (coalesced 256B)
__global__ __launch_bounds__(256)
void scatter_edges(const int* __restrict__ src, const int* __restrict__ dst,
                   const float* __restrict__ x, float* __restrict__ agg, int E) {
    int idx = blockIdx.x * blockDim.x + threadIdx.x;
    int e = idx >> 5;
    int c = (idx & 31) * 2;
    if (e >= E) return;
    int s = src[e], d = dst[e];
    float2 v = *(const float2*)(x + (size_t)s * HDIM + c);
    float* ap = agg + (size_t)d * HDIM + c;
    atomicAdd(ap,     v.x);
    atomicAdd(ap + 1, v.y);
}

// ---------- host launcher ----------
extern "C" void kernel_launch(void* const* d_in, const int* in_sizes, int n_in,
                              void* d_out, int out_size, void* d_ws, size_t ws_size,
                              hipStream_t stream) {
    const float* uF  = (const float*)d_in[0];
    const float* bF  = (const float*)d_in[1];
    const int*   ei  = (const int*)d_in[2];
    const float* Wu1 = (const float*)d_in[3];  const float* bu1 = (const float*)d_in[4];
    const float* Wu2 = (const float*)d_in[5];  const float* bu2 = (const float*)d_in[6];
    const float* Wb1 = (const float*)d_in[7];  const float* bb1 = (const float*)d_in[8];
    const float* Wb2 = (const float*)d_in[9];  const float* bb2 = (const float*)d_in[10];
    const float* Wl  = (const float*)d_in[11]; const float* bl  = (const float*)d_in[12];
    const float* Wr  = (const float*)d_in[13];
    const float* gn  = (const float*)d_in[14]; const float* bt  = (const float*)d_in[15];
    const float* mn  = (const float*)d_in[16]; const float* vr  = (const float*)d_in[17];

    const int* e_src = ei;
    const int* e_dst = ei + NEDGES;
    float* x = (float*)d_out;   // [NNODES, 64] lives in d_out

    // workspace carve (256B aligned)
    size_t off = 0;
    auto carve = [&](size_t bytes) -> char* {
        char* p = (char*)d_ws + off;
        off += (bytes + 255) & ~(size_t)255;
        return p;
    };
    unsigned int* wu1p = (unsigned int*)carve((size_t)HDIM * FIN * 2);
    unsigned int* wb1p = (unsigned int*)carve((size_t)HDIM * FIN * 2);
    unsigned int* wu2p = (unsigned int*)carve((size_t)HDIM * HDIM * 2);
    unsigned int* wb2p = (unsigned int*)carve((size_t)HDIM * HDIM * 2);
    unsigned int* wl0p = (unsigned int*)carve((size_t)HDIM * HDIM * 2);
    unsigned int* wl1p = (unsigned int*)carve((size_t)HDIM * HDIM * 2);
    unsigned int* wr0p = (unsigned int*)carve((size_t)HDIM * HDIM * 2);
    unsigned int* wr1p = (unsigned int*)carve((size_t)HDIM * HDIM * 2);
    float* inv  = (float*)carve((size_t)NNODES * 4);
    float* hbuf = (float*)carve((size_t)NU * HDIM * 4);
    float* tmp  = (float*)carve((size_t)NNODES * HDIM * 4);
    float* agg  = (float*)carve((size_t)NNODES * HDIM * 4);

    // 1) pre-pack weights into WMMA B-fragment bf16 layout
    auto pack = [&](const float* W, unsigned int* Wp, int K) {
        int tiles = 4 * (K >> 5);
        pack_weight<<<(tiles + 7) / 8, 256, 0, stream>>>(W, Wp, K);
    };
    pack(Wu1, wu1p, FIN);  pack(Wb1, wb1p, FIN);
    pack(Wu2, wu2p, HDIM); pack(Wb2, wb2p, HDIM);
    pack(Wl,            wl0p, HDIM); pack(Wl + HDIM * HDIM, wl1p, HDIM);
    pack(Wr,            wr0p, HDIM); pack(Wr + HDIM * HDIM, wr1p, HDIM);

    auto gemm = [&](const float* X, const unsigned int* Wp, float* O, int M, int K,
                    const float* rs, const float* bias, const float* add,
                    const float* g, const float* b2, const float* m2, const float* v2, int relu) {
        int tiles = M / 16;
        gemm_wmma_bf16<<<(tiles + 7) / 8, 256, 0, stream>>>(X, Wp, O, M, K, rs, bias, add,
                                                            g, b2, m2, v2, relu);
    };

    // 2) encoder MLPs: relu(X@W1^T + b1) @ W2^T + b2  -> x rows
    gemm(uF, wu1p, hbuf, NU, FIN, nullptr, bu1, nullptr, nullptr, nullptr, nullptr, nullptr, 1);
    gemm(hbuf, wu2p, x, NU, HDIM, nullptr, bu2, nullptr, nullptr, nullptr, nullptr, nullptr, 0);
    gemm(bF, wb1p, hbuf, NB, FIN, nullptr, bb1, nullptr, nullptr, nullptr, nullptr, nullptr, 1);
    gemm(hbuf, wb2p, x + (size_t)NU * HDIM, NB, HDIM, nullptr, bb2, nullptr,
         nullptr, nullptr, nullptr, nullptr, 0);

    // 3) inverse mean-degree (computed once)
    zero4_kernel<<<(NNODES / 4 + 255) / 256, 256, 0, stream>>>((float4*)inv, NNODES / 4);
    degree_kernel<<<(NEDGES + 255) / 256, 256, 0, stream>>>(e_dst, inv, NEDGES);
    invdeg_kernel<<<(NNODES + 255) / 256, 256, 0, stream>>>(inv, NNODES);

    // 4) SAGE layers: x = relu(BN( (mean_agg)@Wl^T + bl + x@Wr^T ))
    const unsigned int* wlp[2] = {wl0p, wl1p};
    const unsigned int* wrp[2] = {wr0p, wr1p};
    for (int i = 0; i < 2; ++i) {
        gemm(x, wrp[i], tmp, NNODES, HDIM, nullptr, nullptr, nullptr,
             nullptr, nullptr, nullptr, nullptr, 0);                       // tmp = x@Wr^T
        zero4_kernel<<<(NNODES * HDIM / 4 + 255) / 256, 256, 0, stream>>>(
            (float4*)agg, NNODES * HDIM / 4);
        scatter_edges<<<(NEDGES * 32 + 255) / 256, 256, 0, stream>>>(
            e_src, e_dst, x, agg, NEDGES);                                 // agg = sum_j x_j
        gemm(agg, wlp[i], x, NNODES, HDIM, inv, bl + i * HDIM, tmp,
             gn + i * HDIM, bt + i * HDIM, mn + i * HDIM, vr + i * HDIM, 1); // combine+BN+ReLU
    }
}
